// LatentLayer_40673340293237
// MI455X (gfx1250) — matrix operand order repaired and checked by reading, not compiled, not run on hardware
//
#include <hip/hip_runtime.h>
#include <hip/hip_bf16.h>
#include <math.h>

// CDNA5 (gfx1250) fused "tanh -> pairwise sqdist -> min over N -> mean".
// v4: split-bf16 (hi/lo) fp32 emulation on V_WMMA_F32_16X16X32_BF16:
//   a*b ~= ah*bh + ah*bl + al*bh   (al*bl ~ 2^-32, dropped)
// 6 bf16 WMMAs replace 16 f32 WMMAs per 16x16xK=64 tile.
// 4 independent accumulator chains (2 col-tiles x 2 k-chunks) keep the
// WMMA RAW hazard distance >= 4 so the scheduler needs no v_nops.

typedef float v8f __attribute__((ext_vector_type(8)));
typedef __bf16 v4bf __attribute__((ext_vector_type(4)));
typedef __bf16 v8bf __attribute__((ext_vector_type(8)));
typedef __bf16 v16bf __attribute__((ext_vector_type(16)));

#define ZD 64

__device__ __forceinline__ v16bf cat8(v8bf a, v8bf b) {
  return __builtin_shufflevector(a, b, 0, 1, 2, 3, 4, 5, 6, 7,
                                 8, 9, 10, 11, 12, 13, 14, 15);
}

// Order-preserving float <-> uint key mapping (for atomicMin on floats).
__device__ __forceinline__ unsigned fkey(float f) {
  unsigned u = __float_as_uint(f);
  return (u & 0x80000000u) ? ~u : (u | 0x80000000u);
}
__device__ __forceinline__ float funkey(unsigned k) {
  unsigned u = (k & 0x80000000u) ? (k ^ 0x80000000u) : ~k;
  return __uint_as_float(u);
}

__device__ __forceinline__ __bf16 bf_hi(float s) { return (__bf16)s; }
__device__ __forceinline__ __bf16 bf_lo(float s, __bf16 h) {
  return (__bf16)(s - (float)h);
}

// s = -2*tanh(z) split into zh+zl (bf16 hi/lo); zn = ||tanh(z)||^2 per row.
__global__ void prep_z_kernel(const float* __restrict__ z,
                              __bf16* __restrict__ zh,
                              __bf16* __restrict__ zl,
                              float* __restrict__ zn, int N) {
  int row = blockIdx.x * blockDim.x + threadIdx.x;
  if (row >= N) return;
  const float4* src = (const float4*)(z + (size_t)row * ZD);
  float s = 0.f;
#pragma unroll
  for (int i = 0; i < ZD / 4; ++i) {
    float4 v = src[i];
    float t0 = tanhf(v.x), t1 = tanhf(v.y), t2 = tanhf(v.z), t3 = tanhf(v.w);
    s += t0 * t0 + t1 * t1 + t2 * t2 + t3 * t3;
    float s0 = -2.f * t0, s1 = -2.f * t1, s2 = -2.f * t2, s3 = -2.f * t3;
    v4bf h4, l4;
    __bf16 h;
    h = bf_hi(s0); h4[0] = h; l4[0] = bf_lo(s0, h);
    h = bf_hi(s1); h4[1] = h; l4[1] = bf_lo(s1, h);
    h = bf_hi(s2); h4[2] = h; l4[2] = bf_lo(s2, h);
    h = bf_hi(s3); h4[3] = h; l4[3] = bf_lo(s3, h);
    *(v4bf*)(zh + (size_t)row * ZD + i * 4) = h4;
    *(v4bf*)(zl + (size_t)row * ZD + i * 4) = l4;
  }
  zn[row] = s;
}

// e split into eh+el; en = ||e||^2; colmin init to +inf key (every launch).
__global__ void prep_e_kernel(const float* __restrict__ e,
                              __bf16* __restrict__ eh,
                              __bf16* __restrict__ el,
                              float* __restrict__ en,
                              unsigned* __restrict__ colmin, int M) {
  int row = blockIdx.x * blockDim.x + threadIdx.x;
  if (row >= M) return;
  const float4* src = (const float4*)(e + (size_t)row * ZD);
  float s = 0.f;
#pragma unroll
  for (int i = 0; i < ZD / 4; ++i) {
    float4 v = src[i];
    s += v.x * v.x + v.y * v.y + v.z * v.z + v.w * v.w;
    v4bf h4, l4;
    __bf16 h;
    h = bf_hi(v.x); h4[0] = h; l4[0] = bf_lo(v.x, h);
    h = bf_hi(v.y); h4[1] = h; l4[1] = bf_lo(v.y, h);
    h = bf_hi(v.z); h4[2] = h; l4[2] = bf_lo(v.z, h);
    h = bf_hi(v.w); h4[3] = h; l4[3] = bf_lo(v.w, h);
    *(v4bf*)(eh + (size_t)row * ZD + i * 4) = h4;
    *(v4bf*)(el + (size_t)row * ZD + i * 4) = l4;
  }
  en[row] = s;
  colmin[row] = 0xFFFFFFFFu;
}

// Fused split-bf16 WMMA GEMM + column-min.
// Block = 256 threads = 8 wave32: 4 row-groups x 2 column-groups.
// Wave owns TWO 16-column B tiles (hi+lo, both k-chunks register resident),
// streams 16-row A tiles through FOUR accumulator chains, 12 WMMA/row-tile.
__launch_bounds__(256)
__global__ void dist_min_kernel(const __bf16* __restrict__ zh,
                                const __bf16* __restrict__ zl,
                                const float* __restrict__ zn,
                                const __bf16* __restrict__ eh,
                                const __bf16* __restrict__ el,
                                unsigned* __restrict__ colmin,
                                int rowsPerBlock) {
  const int lane = threadIdx.x & 31;
  const int wave = threadIdx.x >> 5;
  const int n = lane & 15;                 // A row / D column index in tile
  const int half = lane >> 4;              // lane half
  const int kg0 = half * 8;                // (unused) A uses contiguous K base kbh
  const int kbh = half * 16;               // B: K base (0..15 | 16..31)
  const int halfoff = half * 8;            // D rows: 0..7 or 8..15
  const int cg = wave & 1;                 // column group (0..1)
  const int rg = wave >> 1;                // row group   (0..3)

  const int colbase = blockIdx.x * 64 + cg * 32;

  // Preload B fragments: [tile 0/1][kchunk 0/1] for hi and lo.
  // Each lane needs 16 contiguous K values of one e-row -> one 32B load.
  v16bf Bh[2][2], Bl[2][2];
#pragma unroll
  for (int t = 0; t < 2; ++t) {
    const size_t roff = (size_t)(colbase + t * 16 + n) * ZD;
#pragma unroll
    for (int kc = 0; kc < 2; ++kc) {
      Bh[t][kc] = *(const v16bf*)(eh + roff + kc * 32 + kbh);
      Bl[t][kc] = *(const v16bf*)(el + roff + kc * 32 + kbh);
    }
  }

  float dmin0[8], dmin1[8];
#pragma unroll
  for (int r = 0; r < 8; ++r) {
    dmin0[r] = __builtin_inff();
    dmin1[r] = __builtin_inff();
  }

  const int rowBeg = blockIdx.y * rowsPerBlock + rg * 16;
  const int rowEnd = blockIdx.y * rowsPerBlock + rowsPerBlock;

  for (int rowbase = rowBeg; rowbase < rowEnd; rowbase += 64) {
    const __bf16* ah = zh + (size_t)(rowbase + n) * ZD;
    const __bf16* al = zl + (size_t)(rowbase + n) * ZD;
    v16bf Ah[2], Al[2];
#pragma unroll
    for (int kc = 0; kc < 2; ++kc) {
      Ah[kc] = *(const v16bf*)(ah + kc * 32 + kbh);
      Al[kc] = *(const v16bf*)(al + kc * 32 + kbh);
    }

    // acc ~= -2 * zt @ e^T via ah*bh + ah*bl + al*bh.
    // 4 independent chains: [col tile][k chunk], each only 3 WMMAs deep.
    v8f acc[2][2] = {{{}, {}}, {{}, {}}};
#pragma unroll
    for (int kc = 0; kc < 2; ++kc) {
      acc[0][kc] = __builtin_amdgcn_wmma_f32_16x16x32_bf16(
          false, Ah[kc], false, Bh[0][kc], (short)0, acc[0][kc], false, false);
      acc[1][kc] = __builtin_amdgcn_wmma_f32_16x16x32_bf16(
          false, Ah[kc], false, Bh[1][kc], (short)0, acc[1][kc], false, false);
    }
#pragma unroll
    for (int kc = 0; kc < 2; ++kc) {
      acc[0][kc] = __builtin_amdgcn_wmma_f32_16x16x32_bf16(
          false, Ah[kc], false, Bl[0][kc], (short)0, acc[0][kc], false, false);
      acc[1][kc] = __builtin_amdgcn_wmma_f32_16x16x32_bf16(
          false, Ah[kc], false, Bl[1][kc], (short)0, acc[1][kc], false, false);
    }
#pragma unroll
    for (int kc = 0; kc < 2; ++kc) {
      acc[0][kc] = __builtin_amdgcn_wmma_f32_16x16x32_bf16(
          false, Al[kc], false, Bh[0][kc], (short)0, acc[0][kc], false, false);
      acc[1][kc] = __builtin_amdgcn_wmma_f32_16x16x32_bf16(
          false, Al[kc], false, Bh[1][kc], (short)0, acc[1][kc], false, false);
    }

    // zn for this lane's 8 D rows (contiguous, 32B aligned)
    const float4* znp = (const float4*)(zn + rowbase + halfoff);
    float4 z0 = znp[0], z1 = znp[1];
    float znv[8] = {z0.x, z0.y, z0.z, z0.w, z1.x, z1.y, z1.z, z1.w};
#pragma unroll
    for (int r = 0; r < 8; ++r) {
      dmin0[r] = fminf(dmin0[r], znv[r] + (acc[0][0][r] + acc[0][1][r]));
      dmin1[r] = fminf(dmin1[r], znv[r] + (acc[1][0][r] + acc[1][1][r]));
    }
  }

  // Reduce 8 D rows in-lane, then across the two lane halves (same column).
  float cm0 = dmin0[0], cm1 = dmin1[0];
#pragma unroll
  for (int r = 1; r < 8; ++r) {
    cm0 = fminf(cm0, dmin0[r]);
    cm1 = fminf(cm1, dmin1[r]);
  }
  cm0 = fminf(cm0, __shfl_xor(cm0, 16, 32));
  cm1 = fminf(cm1, __shfl_xor(cm1, 16, 32));
  if (lane < 16) {
    atomicMin(&colmin[colbase + n], fkey(cm0));
    atomicMin(&colmin[colbase + 16 + n], fkey(cm1));
  }
}

__global__ void final_reduce_kernel(const unsigned* __restrict__ colmin,
                                    const float* __restrict__ en,
                                    float* __restrict__ out, int M) {
  __shared__ float sbuf[256];
  float s = 0.f;
  for (int m = threadIdx.x; m < M; m += 256)
    s += funkey(colmin[m]) + en[m];
  sbuf[threadIdx.x] = s;
  __syncthreads();
  for (int off = 128; off > 0; off >>= 1) {
    if ((int)threadIdx.x < off) sbuf[threadIdx.x] += sbuf[threadIdx.x + off];
    __syncthreads();
  }
  if (threadIdx.x == 0) out[0] = sbuf[0] / (float)M;
}

extern "C" void kernel_launch(void* const* d_in, const int* in_sizes, int n_in,
                              void* d_out, int out_size, void* d_ws, size_t ws_size,
                              hipStream_t stream) {
  (void)n_in; (void)out_size; (void)ws_size;
  const float* z = (const float*)d_in[0];
  const float* e = (const float*)d_in[1];
  float* out = (float*)d_out;
  const int N = in_sizes[0] / ZD;  // 32768
  const int M = in_sizes[1] / ZD;  // 8192

  // Workspace: zh[N*64] zl[N*64] eh[M*64] el[M*64] (bf16) | zn[N] en[M] (f32)
  //            | colmin[M] (u32)   (~10.3 MB total)
  __bf16* zh = (__bf16*)d_ws;
  __bf16* zl = zh + (size_t)N * ZD;
  __bf16* eh = zl + (size_t)N * ZD;
  __bf16* el = eh + (size_t)M * ZD;
  float* zn = (float*)(el + (size_t)M * ZD);
  float* en = zn + N;
  unsigned* colmin = (unsigned*)(en + M);

  prep_z_kernel<<<(N + 255) / 256, 256, 0, stream>>>(z, zh, zl, zn, N);
  prep_e_kernel<<<(M + 255) / 256, 256, 0, stream>>>(e, eh, el, en, colmin, M);

  const int gridY = 16;
  const int rowsPerBlock = N / gridY;  // 2048 rows/block, divisible by 64
  dim3 grid(M / 64, gridY);
  dist_min_kernel<<<grid, 256, 0, stream>>>(zh, zl, zn, eh, el, colmin,
                                            rowsPerBlock);

  final_reduce_kernel<<<1, 256, 0, stream>>>(colmin, en, out, M);
}
